// MaskedCausalAttention_2920577761716
// MI455X (gfx1250) — compile-verified, matrix operand order checked by reading, not run on hardware
//
#include <hip/hip_runtime.h>
#include <hip/hip_bf16.h>

// ---------------------------------------------------------------------------
// Causal multi-head attention for MI455X (gfx1250, wave32, WMMA bf16).
//   B=2, T=2048, C=1024, 16 heads x 64.
// Pipeline: f32->bf16 convert | QKV GEMM (wmma bf16, async double-buffered) |
// flash-attn (wmma bf16, async double-buffered K/V, streaming softmax) |
// out-proj GEMM -> f32.
// ---------------------------------------------------------------------------

typedef __bf16 bf16;
typedef __attribute__((ext_vector_type(16))) __bf16 v16bf;
typedef __attribute__((ext_vector_type(8)))  float  v8f;

#define T_SEQ 2048
#define NH    16
#define HD    64
#define CDIM  1024
#define BATCH 2

static __device__ inline bf16 f2bf(float f) {
  unsigned int u = __builtin_bit_cast(unsigned int, f);
  u += 0x7fffu + ((u >> 16) & 1u);          // round-to-nearest-even
  unsigned short h = (unsigned short)(u >> 16);
  return __builtin_bit_cast(bf16, h);
}

// ISA 7.12.2: 16-bit A/B fragment element idx -> K index (K-tile of 32).
// lanes 0-15 hold K 0..7 / 16..23, lanes 16-31 hold K 8..15 / 24..31.
static __device__ inline int frag_k(int idx, int half) {
  return (idx < 8) ? (half * 8 + idx) : (16 + half * 8 + (idx - 8));
}

static __device__ inline v8f wmma_bf16(v16bf a, v16bf b, v8f c) {
  return __builtin_amdgcn_wmma_f32_16x16x32_bf16(
      /*neg_a=*/false, a, /*neg_b=*/false, b,
      /*c_mod=*/(short)0, c, /*reuse_a=*/false, /*reuse_b=*/false);
}

// GLOBAL_LOAD_ASYNC_TO_LDS_B128 (ASYNCcnt path): VDST = LDS byte address
// (VGPR), VADDR = 64-bit global address.  Generic pointers to __shared__
// carry the LDS offset in their low 32 bits (LDS aperture), so truncation
// yields the DS address.
static __device__ inline void async_load_lds_b128(void* lds_dst, const void* gsrc) {
  unsigned int  l = (unsigned int)(unsigned long long)(uintptr_t)lds_dst;
  unsigned long long g = (unsigned long long)(uintptr_t)gsrc;
  asm volatile("global_load_async_to_lds_b128 %0, %1, off"
               :: "v"(l), "v"(g) : "memory");
}
static __device__ inline void wait_asynccnt_0() {
  asm volatile("s_wait_asynccnt 0x0" ::: "memory");
}
static __device__ inline void wait_asynccnt_2() {
  asm volatile("s_wait_asynccnt 0x2" ::: "memory");
}
static __device__ inline void wait_asynccnt_3() {
  asm volatile("s_wait_asynccnt 0x3" ::: "memory");
}

// ---------------------------------------------------------------------------
// fp32 -> bf16 elementwise convert
// ---------------------------------------------------------------------------
__global__ void __launch_bounds__(256)
cvt_f32_bf16_kernel(const float* __restrict__ in, bf16* __restrict__ out, int n) {
  int i = blockIdx.x * 256 + threadIdx.x;
  if (i < n) out[i] = f2bf(in[i]);
}

// ---------------------------------------------------------------------------
// C[M,N] = A_bf16[M,K] @ B_bf16[K,N] + bias.   Block tile 128x64, 8 waves,
// async double-buffered LDS staging (3 async b128 loads / thread / K-step).
// ---------------------------------------------------------------------------
template <bool OUT_F32>
__global__ void __launch_bounds__(256)
gemm_bf16_kernel(const bf16* __restrict__ A, const bf16* __restrict__ Bm,
                 const float* __restrict__ bias, void* __restrict__ Cout,
                 int M, int N, int K) {
  __shared__ bf16 Atile[2][128][32];    // 16 KB ping-pong
  __shared__ bf16 Btile[2][32][64];     //  8 KB ping-pong

  const int tid  = threadIdx.x;
  const int wave = tid >> 5;
  const int lane = tid & 31;
  const int half = lane >> 4;
  const int ln   = lane & 15;

  const int m0 = blockIdx.y * 128;
  const int n0 = blockIdx.x * 64;

  // Per-thread staging slots
  const int ar = tid >> 1, ac = (tid & 1) * 16;   // A: two 16B chunks
  const int br = tid >> 3, bc = (tid & 7) * 8;    // B: one 16B chunk

  const v8f zf = {0.f, 0.f, 0.f, 0.f, 0.f, 0.f, 0.f, 0.f};
  v8f acc[4] = {zf, zf, zf, zf};

  const int nsteps = K / 32;

  // prologue: stage K-step 0 into buffer 0
  {
    const bf16* ap = &A[(size_t)(m0 + ar) * K + ac];
    async_load_lds_b128(&Atile[0][ar][ac],     ap);
    async_load_lds_b128(&Atile[0][ar][ac + 8], ap + 8);
    async_load_lds_b128(&Btile[0][br][bc], &Bm[(size_t)br * N + n0 + bc]);
  }

  for (int it = 0; it < nsteps; ++it) {
    const int buf = it & 1;
    if (it + 1 < nsteps) {   // stage next K-step into the other buffer
      const int kk = (it + 1) * 32;
      const bf16* ap = &A[(size_t)(m0 + ar) * K + kk + ac];
      async_load_lds_b128(&Atile[buf ^ 1][ar][ac],     ap);
      async_load_lds_b128(&Atile[buf ^ 1][ar][ac + 8], ap + 8);
      async_load_lds_b128(&Btile[buf ^ 1][br][bc],
                          &Bm[(size_t)(kk + br) * N + n0 + bc]);
      wait_asynccnt_3();     // in-order: current step's 3 loads have landed
    } else {
      wait_asynccnt_0();
    }
    __syncthreads();         // all threads' chunks of this step visible

    v16bf a;
#pragma unroll
    for (int idx = 0; idx < 16; ++idx)
      a[idx] = Atile[buf][wave * 16 + ln][frag_k(idx, half)];

#pragma unroll
    for (int dt = 0; dt < 4; ++dt) {
      v16bf b;
#pragma unroll
      for (int idx = 0; idx < 16; ++idx)
        b[idx] = Btile[buf][frag_k(idx, half)][dt * 16 + ln];
      acc[dt] = wmma_bf16(a, b, acc[dt]);
    }

    __syncthreads();         // buffer free before it is re-staged at it+1
  }

#pragma unroll
  for (int g = 0; g < 8; ++g) {
    int row = m0 + wave * 16 + g + 8 * half;           // D layout: m = g + 8*half
#pragma unroll
    for (int dt = 0; dt < 4; ++dt) {
      int col  = n0 + dt * 16 + ln;                     // D layout: n = lane%16
      float v  = acc[dt][g] + bias[col];
      if (OUT_F32) ((float*)Cout)[(size_t)row * N + col] = v;
      else         ((bf16*)Cout)[(size_t)row * N + col] = f2bf(v);
    }
  }
}

// ---------------------------------------------------------------------------
// Flash attention: one workgroup = (batch, head, 128 query rows); 8 waves x 16
// rows.  Double-buffered async K/V staging; streaming softmax in fp32.
// ---------------------------------------------------------------------------
__global__ void __launch_bounds__(256)
flash_attn_kernel(const bf16* __restrict__ q, const bf16* __restrict__ k,
                  const bf16* __restrict__ v, bf16* __restrict__ o) {
  __shared__ bf16 Kld[2][32][HD];       // 8 KB (ping-pong)
  __shared__ bf16 Vld[2][32][HD];       // 8 KB (ping-pong)
  __shared__ bf16 Pld[8][16][32];       // 8 KB, wave-private P staging

  const int tid  = threadIdx.x;
  const int wave = tid >> 5;
  const int lane = tid & 31;
  const int half = lane >> 4;
  const int ln   = lane & 15;

  const int rb = blockIdx.x;            // 128-row strip
  const int h  = blockIdx.y;            // head
  const int b  = blockIdx.z;            // batch
  const int qr0 = rb * 128 + wave * 16; // first query row of this wave
  const size_t rowbase = (size_t)b * T_SEQ;

  // Per-thread staging slot: 8 bf16 (16 B) of K and of V per 32-key block.
  const int ld_key = tid >> 3;          // 0..31
  const int ld_d   = (tid & 7) * 8;     // 0,8,...,56

  // Q fragments (16x64 -> two 16x32 A frags), loaded straight in A layout.
  v16bf qa0, qa1;
  {
    const bf16* qp = q + (rowbase + qr0 + ln) * CDIM + h * HD;
#pragma unroll
    for (int idx = 0; idx < 16; ++idx) {
      int kc = frag_k(idx, half);
      qa0[idx] = qp[kc];
      qa1[idx] = qp[32 + kc];
    }
  }

  const v8f zf = {0.f, 0.f, 0.f, 0.f, 0.f, 0.f, 0.f, 0.f};
  v8f acc[4] = {zf, zf, zf, zf};        // O accumulator 16x64 fp32
  float rmax[8], rsum[8];
#pragma unroll
  for (int g = 0; g < 8; ++g) { rmax[g] = -3.0e38f; rsum[g] = 0.f; }

  const int nkb = (rb + 1) * 4;         // causal: key blocks of 32

  {   // prologue: async-stage block 0 into buffer 0
    const size_t src = (rowbase + ld_key) * CDIM + h * HD + ld_d;
    async_load_lds_b128(&Kld[0][ld_key][ld_d], &k[src]);
    async_load_lds_b128(&Vld[0][ld_key][ld_d], &v[src]);
  }

  for (int kb = 0; kb < nkb; ++kb) {
    const int buf = kb & 1;
    if (kb + 1 < nkb) {   // stage next block into the other buffer
      const size_t src = (rowbase + (size_t)(kb + 1) * 32 + ld_key) * CDIM
                         + h * HD + ld_d;
      async_load_lds_b128(&Kld[buf ^ 1][ld_key][ld_d], &k[src]);
      async_load_lds_b128(&Vld[buf ^ 1][ld_key][ld_d], &v[src]);
      wait_asynccnt_2();  // in-order: current block's 2 loads have landed
    } else {
      wait_asynccnt_0();
    }
    __syncthreads();      // all waves' chunks of block kb visible

    if (kb * 32 <= qr0 + 15) {          // wave-uniform causal skip
      // S = Q (16x64) x K^T (64x32): two 16x16 D tiles, 2 k-steps each
      v8f c0 = zf, c1 = zf;
      v16bf b00, b01, b10, b11;
#pragma unroll
      for (int idx = 0; idx < 16; ++idx) {
        int kc = frag_k(idx, half);     // feature-dim index within 32-chunk
        b00[idx] = Kld[buf][ln][kc];         // keys 0..15,  d 0..31
        b01[idx] = Kld[buf][ln][32 + kc];    // keys 0..15,  d 32..63
        b10[idx] = Kld[buf][16 + ln][kc];    // keys 16..31, d 0..31
        b11[idx] = Kld[buf][16 + ln][32 + kc];
      }
      c0 = wmma_bf16(qa0, b00, c0);
      c0 = wmma_bf16(qa1, b01, c0);
      c1 = wmma_bf16(qa0, b10, c1);
      c1 = wmma_bf16(qa1, b11, c1);

      // scale + causal mask + streaming softmax (rows spread over lane halves)
#pragma unroll
      for (int g = 0; g < 8; ++g) {
        int row  = qr0 + g + 8 * half;
        int col0 = kb * 32 + ln;
        int col1 = col0 + 16;
        float s0 = (col0 <= row) ? c0[g] * 0.125f : -3.0e38f;
        float s1 = (col1 <= row) ? c1[g] * 0.125f : -3.0e38f;
        float mx = fmaxf(s0, s1);
#pragma unroll
        for (int off = 1; off < 16; off <<= 1)
          mx = fmaxf(mx, __shfl_xor(mx, off, 32));
        mx = fmaxf(mx, rmax[g]);
        float corr = __expf(rmax[g] - mx);
        rmax[g] = mx;
        float e0 = __expf(s0 - mx);
        float e1 = __expf(s1 - mx);
        float es = e0 + e1;
#pragma unroll
        for (int off = 1; off < 16; off <<= 1)
          es += __shfl_xor(es, off, 32);
        rsum[g] = rsum[g] * corr + es;
        acc[0][g] *= corr; acc[1][g] *= corr;
        acc[2][g] *= corr; acc[3][g] *= corr;
        // stage P (D layout) into wave-private LDS
        Pld[wave][g + 8 * half][ln]      = f2bf(e0);
        Pld[wave][g + 8 * half][16 + ln] = f2bf(e1);
      }

      // reload P as 16x32 A fragment (same-wave LDS ops are in-order)
      v16bf pa;
#pragma unroll
      for (int idx = 0; idx < 16; ++idx)
        pa[idx] = Pld[wave][ln][frag_k(idx, half)];

      // O += P (16x32) x V (32x64)
#pragma unroll
      for (int dt = 0; dt < 4; ++dt) {
        v16bf vb;
#pragma unroll
        for (int idx = 0; idx < 16; ++idx)
          vb[idx] = Vld[buf][frag_k(idx, half)][dt * 16 + ln];
        acc[dt] = wmma_bf16(pa, vb, acc[dt]);
      }
    }

    __syncthreads();      // buffer `buf` free before it is re-staged at kb+1
  }

  // epilogue: normalize and store bf16 [B,T,C] with head's 64-col slice
#pragma unroll
  for (int g = 0; g < 8; ++g) {
    int row   = qr0 + g + 8 * half;
    float inv = 1.f / rsum[g];
    bf16* op  = o + (rowbase + row) * CDIM + h * HD;
#pragma unroll
    for (int dt = 0; dt < 4; ++dt)
      op[dt * 16 + ln] = f2bf(acc[dt][g] * inv);
  }
}

// ---------------------------------------------------------------------------
// Host launcher
// ---------------------------------------------------------------------------
extern "C" void kernel_launch(void* const* d_in, const int* in_sizes, int n_in,
                              void* d_out, int out_size, void* d_ws, size_t ws_size,
                              hipStream_t stream) {
  (void)in_sizes; (void)n_in; (void)out_size; (void)ws_size;

  const float* src = (const float*)d_in[0];
  // d_in[1] = mask, unused (causal)
  const float* Wq = (const float*)d_in[2];
  const float* bq = (const float*)d_in[3];
  const float* Wk = (const float*)d_in[4];
  const float* bk = (const float*)d_in[5];
  const float* Wv = (const float*)d_in[6];
  const float* bv = (const float*)d_in[7];
  const float* Wo = (const float*)d_in[8];
  const float* bo = (const float*)d_in[9];

  const int M     = BATCH * T_SEQ;       // 4096
  const int SRC_N = M * CDIM;            // 4,194,304
  const int W_N   = CDIM * CDIM;         // 1,048,576

  bf16* wsb   = (bf16*)d_ws;             // 48 MB total in bf16 scratch
  bf16* srcb  = wsb;
  bf16* wqb   = srcb + SRC_N;
  bf16* wkb   = wqb + W_N;
  bf16* wvb   = wkb + W_N;
  bf16* wob   = wvb + W_N;
  bf16* qb    = wob + W_N;
  bf16* kbuf  = qb + SRC_N;
  bf16* vbuf  = kbuf + SRC_N;
  bf16* attnb = vbuf + SRC_N;

  // 1) fp32 -> bf16 conversions
  cvt_f32_bf16_kernel<<<SRC_N / 256, 256, 0, stream>>>(src, srcb, SRC_N);
  cvt_f32_bf16_kernel<<<W_N / 256, 256, 0, stream>>>(Wq, wqb, W_N);
  cvt_f32_bf16_kernel<<<W_N / 256, 256, 0, stream>>>(Wk, wkb, W_N);
  cvt_f32_bf16_kernel<<<W_N / 256, 256, 0, stream>>>(Wv, wvb, W_N);
  cvt_f32_bf16_kernel<<<W_N / 256, 256, 0, stream>>>(Wo, wob, W_N);

  // 2) QKV projections (WMMA GEMM, async double-buffered -> bf16)
  dim3 ggrid(CDIM / 64, M / 128);
  gemm_bf16_kernel<false><<<ggrid, 256, 0, stream>>>(srcb, wqb, bq, qb,   M, CDIM, CDIM);
  gemm_bf16_kernel<false><<<ggrid, 256, 0, stream>>>(srcb, wkb, bk, kbuf, M, CDIM, CDIM);
  gemm_bf16_kernel<false><<<ggrid, 256, 0, stream>>>(srcb, wvb, bv, vbuf, M, CDIM, CDIM);

  // 3) flash attention (causal, streaming softmax, async double-buffered K/V)
  dim3 agrid(T_SEQ / 128, NH, BATCH);
  flash_attn_kernel<<<agrid, 256, 0, stream>>>(qb, kbuf, vbuf, attnb);

  // 4) output projection (WMMA GEMM, async double-buffered -> fp32 d_out)
  gemm_bf16_kernel<true><<<ggrid, 256, 0, stream>>>(attnb, wob, bo, d_out, M, CDIM, CDIM);
}
